// SRADiffAttn_84396107366633
// MI455X (gfx1250) — matrix operand order repaired and checked by reading, not compiled, not run on hardware
//
#include <hip/hip_runtime.h>

// ---------------------------------------------------------------------------
// Differential attention for MI455X (gfx1250, wave32, WMMA).
// B=2 N=2048 D=1024 H=16 HKV=4 HD=32 DV=64.
// All matmuls run on v_wmma_f32_16x16x32_bf16. Attention is flash-style
// (online softmax for both differential components, combined in epilogue with
// RMSNorm), so the [B,32,N,N] weights (1 GB) are never materialized.
// Row reductions use DPP16 butterflies (VALU) instead of ds_bpermute.
// ---------------------------------------------------------------------------

typedef __attribute__((ext_vector_type(16))) __bf16 v16bf;
typedef __attribute__((ext_vector_type(8)))  __bf16 v8bf;
typedef __attribute__((ext_vector_type(8)))  float  v8f;

#define BDIM      2
#define NDIM      2048
#define DDIM      1024
#define HHEADS    16
#define SCALE_Q   0.17677669529663687f     // 32^-0.5
#define LAMBDA_I  0.7008206670670481f      // 0.8 - 0.6*exp(-0.3*6)
#define ONE_M_LI  0.2991793329329519f
#define RMS_EPS_F 1e-5f

__device__ __forceinline__ v8f wmma_bf16(v16bf a, v16bf b, v8f c) {
    return __builtin_amdgcn_wmma_f32_16x16x32_bf16(
        /*neg_a=*/false, a, /*neg_b=*/false, b,
        /*c_mod=*/(short)0, c, /*reuse_a=*/false, /*reuse_b=*/false);
}

// --- DPP16 cross-lane helpers (reduction over the 16-lane wave32 half) -----
// quad_perm(1,0,3,2)=0xB1, quad_perm(2,3,0,1)=0x4E,
// ROW_HALF_MIRROR=0x141, ROW_MIRROR=0x140.
template <int CTRL>
__device__ __forceinline__ float dpp_mov(float x) {
    int xi = __builtin_bit_cast(int, x);
    return __builtin_bit_cast(
        float, __builtin_amdgcn_update_dpp(xi, xi, CTRL, 0xf, 0xf, true));
}
__device__ __forceinline__ float red16_max(float x) {
    x = fmaxf(x, dpp_mov<0xB1>(x));
    x = fmaxf(x, dpp_mov<0x4E>(x));
    x = fmaxf(x, dpp_mov<0x141>(x));
    x = fmaxf(x, dpp_mov<0x140>(x));
    return x;
}
__device__ __forceinline__ float red16_sum(float x) {
    x += dpp_mov<0xB1>(x);
    x += dpp_mov<0x4E>(x);
    x += dpp_mov<0x141>(x);
    x += dpp_mov<0x140>(x);
    return x;
}

// A fragment (16x32 bf16, row-major, leading dim ld):
// lane L holds row m = L&15; K = (L>>4)*8 .. +8 and (L>>4)*8+16 .. +8.
__device__ __forceinline__ v16bf load_frag_A(const __bf16* p, int ld, int lane) {
    const int m = lane & 15, hi = lane >> 4;
    const __bf16* q = p + (size_t)m * ld + hi * 8;
    v8bf lo = *(const v8bf*)(q);
    v8bf hv = *(const v8bf*)(q + 16);
    v16bf r;
#pragma unroll
    for (int i = 0; i < 8; ++i) { r[i] = lo[i]; r[i + 8] = hv[i]; }
    return r;
}

// B fragment (32x16): B[k][n] = src[n][k] with src row-major, leading dim ld.
// lane L holds col n = L&15; K = (L>>4)*8 .. +8 and (L>>4)*8+16 .. +8.
__device__ __forceinline__ v16bf load_frag_B(const __bf16* p, int ld, int lane) {
    const int n = lane & 15, hi = lane >> 4;
    const __bf16* q = p + (size_t)n * ld + hi * 8;
    v8bf lo = *(const v8bf*)(q);
    v8bf hv = *(const v8bf*)(q + 16);
    v16bf r;
#pragma unroll
    for (int i = 0; i < 8; ++i) { r[i] = lo[i]; r[i + 8] = hv[i]; }
    return r;
}

// ---------------------------------------------------------------------------
__global__ void cvt_f32_to_bf16(const float* __restrict__ src,
                                __bf16* __restrict__ dst, int n) {
    int i = blockIdx.x * blockDim.x + threadIdx.x;
    if (i < n) dst[i] = (__bf16)src[i];
}

// ---------------------------------------------------------------------------
// Generic bf16 GEMM: Y = A[M=4096,K=1024] @ W[Nc,K]^T, epilogue per MODE.
// Block: 256 threads = 8 waves in 4x2; wave tile 32x32; WG tile 128x64.
// MODE 0: q  -> bf16 [B,32,N,32] head-major, scaled by hd^-0.5
// MODE 1: k  -> bf16 [B, 8,N,32]
// MODE 2: v  -> bf16 [B, 4,N,64]
// MODE 3: o  -> f32  [M,1024]
template <int MODE>
__global__ __launch_bounds__(256)
void gemm_bf16(const __bf16* __restrict__ A, const __bf16* __restrict__ W,
               void* __restrict__ outp) {
    constexpr int K = DDIM;
    const int lane = threadIdx.x & 31, wave = threadIdx.x >> 5;
    const int m0 = blockIdx.x * 128 + (wave & 3) * 32;
    const int n0 = blockIdx.y * 64 + (wave >> 2) * 32;
    v8f zero = {};
    v8f c00 = zero, c01 = zero, c10 = zero, c11 = zero;
    const __bf16* a0p = A + (size_t)m0 * K;
    const __bf16* a1p = A + (size_t)(m0 + 16) * K;
    const __bf16* b0p = W + (size_t)n0 * K;
    const __bf16* b1p = W + (size_t)(n0 + 16) * K;
    for (int k0 = 0; k0 < K; k0 += 32) {
        v16bf a0 = load_frag_A(a0p + k0, K, lane);
        v16bf a1 = load_frag_A(a1p + k0, K, lane);
        v16bf b0 = load_frag_B(b0p + k0, K, lane);
        v16bf b1 = load_frag_B(b1p + k0, K, lane);
        c00 = wmma_bf16(a0, b0, c00);
        c01 = wmma_bf16(a0, b1, c01);
        c10 = wmma_bf16(a1, b0, c10);
        c11 = wmma_bf16(a1, b1, c11);
    }
    const int hi = lane >> 4, col = lane & 15;
    auto store_tile = [&](v8f c, int mbase, int nbase) {
#pragma unroll
        for (int r = 0; r < 8; ++r) {
            const int m = mbase + r + 8 * hi;
            const int n = nbase + col;
            const float v = c[r];
            const int b = m >> 11, row = m & (NDIM - 1);
            if constexpr (MODE == 0) {
                ((__bf16*)outp)[((((size_t)b * 32 + (n >> 5)) * NDIM + row) << 5) + (n & 31)]
                    = (__bf16)(v * SCALE_Q);
            } else if constexpr (MODE == 1) {
                ((__bf16*)outp)[((((size_t)b * 8 + (n >> 5)) * NDIM + row) << 5) + (n & 31)]
                    = (__bf16)v;
            } else if constexpr (MODE == 2) {
                ((__bf16*)outp)[((((size_t)b * 4 + (n >> 6)) * NDIM + row) << 6) + (n & 63)]
                    = (__bf16)v;
            } else {
                ((float*)outp)[(size_t)m * DDIM + n] = v;
            }
        }
    };
    store_tile(c00, m0, n0);
    store_tile(c01, m0, n0 + 16);
    store_tile(c10, m0 + 16, n0);
    store_tile(c11, m0 + 16, n0 + 16);
}

// ---------------------------------------------------------------------------
// Online softmax update for one component over a 32-key block.
// sa/sb: C-layout scores (keys 0-15 / 16-31). Stats/rescale per row r+8*hi,
// row reductions = DPP16 butterflies within each wave32 half (pure VALU).
// Writes exp(P) bf16 to wave-private LDS in 16x32 row-major for A-frag reload.
__device__ __forceinline__ void online_update(v8f sa, v8f sb, float* mx,
                                              float* sum, v8f* o,
                                              __bf16* pbuf, int lane) {
    const int hi = lane >> 4, col = lane & 15;
#pragma unroll
    for (int r = 0; r < 8; ++r) {
        const float rm = red16_max(fmaxf(sa[r], sb[r]));
        const float mn = fmaxf(mx[r], rm);
        const float corr = __expf(mx[r] - mn);
        mx[r] = mn;
        const float pa = __expf(sa[r] - mn);
        const float pb = __expf(sb[r] - mn);
        const float rs = red16_sum(pa + pb);
        sum[r] = sum[r] * corr + rs;
#pragma unroll
        for (int s = 0; s < 4; ++s) o[s][r] *= corr;
        const int row = r + 8 * hi;
        pbuf[row * 32 + col] = (__bf16)pa;
        pbuf[row * 32 + col + 16] = (__bf16)pb;
    }
}

// ---------------------------------------------------------------------------
// Flash differential attention. Grid: (B*H, N/64); 128 threads = 4 waves,
// each wave owns 16 query rows. Both differential components of head h use
// kv head h>>1 (repeat_interleave(4) of 8 k-heads) and v head h>>2.
__global__ __launch_bounds__(128)
void diff_flash_attn(const __bf16* __restrict__ qb, const __bf16* __restrict__ kb,
                     const __bf16* __restrict__ vb,
                     const float* __restrict__ lq1, const float* __restrict__ lk1,
                     const float* __restrict__ lq2, const float* __restrict__ lk2,
                     const float* __restrict__ rmsw, __bf16* __restrict__ attnbuf) {
    constexpr int N = NDIM;
    __shared__ alignas(16) __bf16 kS[32 * 32];        // K block [key][hd]
    __shared__ alignas(16) __bf16 vtS[64 * 40];       // V^T block [vdim][key], padded
    __shared__ alignas(16) __bf16 pS[4][2][16 * 32];  // per-wave P staging
    const int b = blockIdx.x >> 4, h = blockIdx.x & 15;
    const int wave = threadIdx.x >> 5, lane = threadIdx.x & 31;
    const int hi = lane >> 4, col = lane & 15;
    const int qrow = blockIdx.y * 64 + wave * 16;
    const __bf16* Q1 = qb + (((size_t)b * 32 + 2 * h) * N + qrow) * 32;
    const __bf16* Q2 = qb + (((size_t)b * 32 + 2 * h + 1) * N + qrow) * 32;
    const __bf16* Kp = kb + ((size_t)b * 8 + (h >> 1)) * N * 32;
    const __bf16* Vp = vb + ((size_t)b * 4 + (h >> 2)) * N * 64;

    const v16bf qa1 = load_frag_A(Q1, 32, lane);
    const v16bf qa2 = load_frag_A(Q2, 32, lane);

    float mx1[8], mx2[8], sum1[8], sum2[8];
    v8f zero = {};
    v8f o1[4], o2[4];
#pragma unroll
    for (int r = 0; r < 8; ++r) { mx1[r] = mx2[r] = -3.0e38f; sum1[r] = sum2[r] = 0.f; }
#pragma unroll
    for (int s = 0; s < 4; ++s) { o1[s] = zero; o2[s] = zero; }

    for (int j = 0; j < N; j += 32) {
        __syncthreads();
        // stage K block: 1024 bf16, 8 per thread, b128 both sides
        *(v8bf*)&kS[threadIdx.x * 8] = *(const v8bf*)&Kp[(size_t)j * 32 + threadIdx.x * 8];
        // stage V block transposed: thread -> one 16-wide row segment
        {
            const int key = threadIdx.x >> 2, c0 = (threadIdx.x & 3) * 16;
            v8bf x0 = *(const v8bf*)&Vp[((size_t)(j + key)) * 64 + c0];
            v8bf x1 = *(const v8bf*)&Vp[((size_t)(j + key)) * 64 + c0 + 8];
#pragma unroll
            for (int i = 0; i < 8; ++i) {
                vtS[(c0 + i) * 40 + key] = x0[i];
                vtS[(c0 + 8 + i) * 40 + key] = x1[i];
            }
        }
        __syncthreads();
        if (j + 32 < N) {  // global_prefetch_b8 next K/V blocks
            __builtin_prefetch(&Kp[(size_t)(j + 32) * 32], 0, 1);
            __builtin_prefetch(&Vp[(size_t)(j + 32) * 64], 0, 1);
        }
        // S = Q K^T (shared K head for both components)
        v16bf kf0 = load_frag_B(kS, 32, lane);
        v16bf kf1 = load_frag_B(kS + 16 * 32, 32, lane);
        v8f s1a = wmma_bf16(qa1, kf0, zero), s1b = wmma_bf16(qa1, kf1, zero);
        v8f s2a = wmma_bf16(qa2, kf0, zero), s2b = wmma_bf16(qa2, kf1, zero);

        online_update(s1a, s1b, mx1, sum1, o1, pS[wave][0], lane);
        online_update(s2a, s2b, mx2, sum2, o2, pS[wave][1], lane);
        __builtin_amdgcn_wave_barrier();
        asm volatile("s_wait_dscnt 0" ::: "memory");  // wave-private LDS RAW

        v16bf p1 = load_frag_A(pS[wave][0], 32, lane);
        v16bf p2 = load_frag_A(pS[wave][1], 32, lane);
#pragma unroll
        for (int s = 0; s < 4; ++s) {
            v16bf vf = load_frag_B(&vtS[s * 16 * 40], 40, lane);
            o1[s] = wmma_bf16(p1, vf, o1[s]);
            o2[s] = wmma_bf16(p2, vf, o2[s]);
        }
    }

    // lambda_full = exp(lq1.lk1) - exp(lq2.lk2) + lambda_init (uniform, cached)
    float d1 = 0.f, d2 = 0.f;
#pragma unroll
    for (int i = 0; i < 32; ++i) { d1 += lq1[i] * lk1[i]; d2 += lq2[i] * lk2[i]; }
    const float lam = __expf(d1) - __expf(d2) + LAMBDA_I;

    float outv[4][8];
#pragma unroll
    for (int s = 0; s < 4; ++s)
#pragma unroll
        for (int r = 0; r < 8; ++r)
            outv[s][r] = o1[s][r] / sum1[r] - lam * o2[s][r] / sum2[r];

#pragma unroll
    for (int r = 0; r < 8; ++r) {
        float ss = 0.f;
#pragma unroll
        for (int s = 0; s < 4; ++s) ss += outv[s][r] * outv[s][r];
        ss = red16_sum(ss);
        const float inv = rsqrtf(ss * (1.f / 64.f) + RMS_EPS_F);
        const size_t rowoff = ((size_t)b * N + qrow + r + 8 * hi) * DDIM + (size_t)h * 64;
#pragma unroll
        for (int s = 0; s < 4; ++s) {
            const float w = rmsw[s * 16 + col];
            attnbuf[rowoff + s * 16 + col] = (__bf16)(outv[s][r] * inv * w * ONE_M_LI);
        }
    }
}

// ---------------------------------------------------------------------------
extern "C" void kernel_launch(void* const* d_in, const int* in_sizes, int n_in,
                              void* d_out, int out_size, void* d_ws, size_t ws_size,
                              hipStream_t stream) {
    (void)in_sizes; (void)n_in; (void)out_size; (void)ws_size;
    const float* x   = (const float*)d_in[0];
    const float* Wq  = (const float*)d_in[1];
    const float* Wk  = (const float*)d_in[2];
    const float* Wv  = (const float*)d_in[3];
    const float* Wo  = (const float*)d_in[4];
    const float* lq1 = (const float*)d_in[5];
    const float* lk1 = (const float*)d_in[6];
    const float* lq2 = (const float*)d_in[7];
    const float* lk2 = (const float*)d_in[8];
    const float* rmw = (const float*)d_in[9];
    float* out = (float*)d_out;

    char* ws = (char*)d_ws;
    size_t off = 0;
    auto carve = [&](size_t elems) {
        void* p = ws + off;
        off = (off + elems * sizeof(__bf16) + 255) & ~(size_t)255;
        return (__bf16*)p;
    };
    const size_t nX = (size_t)BDIM * NDIM * DDIM;   // 4M
    __bf16* xb  = carve(nX);
    __bf16* wqb = carve((size_t)DDIM * DDIM);
    __bf16* wkb = carve((size_t)256 * DDIM);
    __bf16* wvb = carve((size_t)256 * DDIM);
    __bf16* wob = carve((size_t)DDIM * DDIM);
    __bf16* q_b = carve(nX);                        // [B,32,N,32]
    __bf16* k_b = carve((size_t)BDIM * 8 * NDIM * 32);
    __bf16* v_b = carve((size_t)BDIM * 4 * NDIM * 64);
    __bf16* att = carve(nX);                        // [B,N,D] post-RMSNorm

    auto cvt = [&](const float* s, __bf16* d, int n) {
        cvt_f32_to_bf16<<<(n + 255) / 256, 256, 0, stream>>>(s, d, n);
    };
    cvt(x,  xb,  (int)nX);
    cvt(Wq, wqb, DDIM * DDIM);
    cvt(Wk, wkb, 256 * DDIM);
    cvt(Wv, wvb, 256 * DDIM);
    cvt(Wo, wob, DDIM * DDIM);

    gemm_bf16<0><<<dim3(32, 16), 256, 0, stream>>>(xb, wqb, q_b);
    gemm_bf16<1><<<dim3(32, 4),  256, 0, stream>>>(xb, wkb, k_b);
    gemm_bf16<2><<<dim3(32, 4),  256, 0, stream>>>(xb, wvb, v_b);

    diff_flash_attn<<<dim3(BDIM * HHEADS, NDIM / 64), 128, 0, stream>>>(
        q_b, k_b, v_b, lq1, lk1, lq2, lk2, rmw, att);

    gemm_bf16<3><<<dim3(32, 16), 256, 0, stream>>>(att, wob, out);
}
